// EnhancedGATNet_89017492177271
// MI455X (gfx1250) — matrix operand order repaired and checked by reading, not compiled, not run on hardware
//
#include <hip/hip_runtime.h>
#include <hip/hip_bf16.h>
#include <math.h>

#define NN   100000
#define EE   640000
#define HIDN 128

typedef __attribute__((ext_vector_type(16))) _Float16 v16h;
typedef __attribute__((ext_vector_type(8)))  float    v8f;
typedef __attribute__((ext_vector_type(4)))  int      v4i;

// CDNA5 async global->LDS path (ASYNCcnt), guarded so either toolchain compiles.
#if __has_builtin(__builtin_amdgcn_global_load_async_to_lds_b128) && __has_builtin(__builtin_amdgcn_s_wait_asynccnt)
#define HAVE_ASYNC 1
#define GAS __attribute__((address_space(1)))
#define LAS __attribute__((address_space(3)))
#else
#define HAVE_ASYNC 0
#endif

// ---- ordered-int mapping for f32 atomicMax (monotonic over all finite floats) ----
__device__ __forceinline__ int f2o(float f) {
    int i = __float_as_int(f);
    return (i >= 0) ? i : (i ^ 0x7fffffff);
}
__device__ __forceinline__ float o2f(int o) {
    return __int_as_float((o >= 0) ? o : (o ^ 0x7fffffff));
}
#define NEG_ORD ((int)0x807fffff)   // f2o(-inf)

// =====================================================================
// WMMA GEMM: out[rows x NO] = A[rows x K] @ W[K x NO] (+ bias)
// block = 16 rows ; one wave per 16 output columns (NW = NO/16 waves)
// A tile: raw f32 rows in LDS (async global->LDS copy when available).
// W tile: f16 in LDS, pre-swizzled into per-lane fragment order so a
//         B fragment is one aligned 32-byte read (2x ds_load_b128):
//         Wsw[n*K + kb + lh*16 + kk]  <=  W[k][n], k = kb + lh*16 + kk
// =====================================================================
template<int K, int NO>
__global__ __launch_bounds__(32 * (NO / 16))
void gemm_wmma(const float* __restrict__ A, const float* __restrict__ W,
               const float* __restrict__ bias, float* __restrict__ out, int rows)
{
    constexpr int NW = NO / 16;
    constexpr int NT = 32 * NW;
    __shared__ __align__(16) float    As[16 * K];
    __shared__ __align__(16) _Float16 Wsw[K * NO];

    const int tid  = threadIdx.x;
    const int row0 = blockIdx.x * 16;

    // ---- stage A tile (16 x K, raw f32) ----
#if HAVE_ASYNC
    if (row0 + 16 <= rows) {
        // full tile: per-lane async 16B copies, tracked by ASYNCcnt
        for (int idx = tid; idx < 4 * K; idx += NT) {   // (16*K f32)/4 per transfer
            const int r = idx / (K / 4), q = idx - r * (K / 4);
            __builtin_amdgcn_global_load_async_to_lds_b128(
                (GAS v4i*)(A + (size_t)(row0 + r) * K + q * 4),
                (LAS v4i*)(As + r * K + q * 4), 0, 0);
        }
        __builtin_amdgcn_s_wait_asynccnt(0);
    } else
#endif
    {
        for (int idx = tid; idx < 16 * K; idx += NT) {
            const int r = idx / K, k = idx - r * K;
            const int gr = row0 + r;
            As[idx] = (gr < rows) ? A[(size_t)gr * K + k] : 0.0f;
        }
    }

    // ---- stage W (K x NO) as f16, swizzled to fragment order ----
    for (int idx = tid; idx < K * NO; idx += NT) {
        const int k = idx / NO, n = idx - k * NO;       // coalesced global read
        const int dst = ((n * (K / 32) + (k >> 5)) * 2 + ((k >> 4) & 1)) * 16 + (k & 15);
        Wsw[dst] = (_Float16)W[idx];
    }
    __syncthreads();

    const int wave = tid >> 5;
    const int lane = tid & 31;
    const int lh   = lane >> 4;   // lane half: 0 (lanes 0-15) / 1 (lanes 16-31)
    const int lm   = lane & 15;
    const int n0   = wave * 16;

    const float*    Arow = &As[lm * K];
    const _Float16* Bcol = &Wsw[(n0 + lm) * K];

    v8f acc = {};
    for (int kb = 0; kb < K; kb += 32) {
        v16h a;
#pragma unroll
        for (int t = 0; t < 16; ++t) {
            const int j = t >> 1, sub = t & 1;
            // A (16x32 f16, MxK) contiguous-K layout: k = kb + lh*16 + t (matches B)
            const int ka = kb + lh * 16 + t; (void)j; (void)sub;
            a[t] = (_Float16)Arow[ka];
        }
        const v16h b = *(const v16h*)&Bcol[kb + lh * 16];   // 32B aligned -> ds_load_b128 x2
        acc = __builtin_amdgcn_wmma_f32_16x16x32_f16(
            /*neg_a=*/false, a, /*neg_b=*/false, b,
            /*c_mod=*/(short)0, acc, /*reuse_a=*/false, /*reuse_b=*/false);
    }

    const int n  = n0 + lm;
    const float bv = bias ? bias[n] : 0.0f;
#pragma unroll
    for (int r = 0; r < 8; ++r) {
        const int m  = r + lh * 8;        // C/D layout: VGPR r -> M=r (lo half) / 8+r (hi half)
        const int gr = row0 + m;
        if (gr < rows) out[(size_t)gr * NO + n] = acc[r] + bv;
    }
}

// =====================================================================
// Per-layer init: zero accumulators, set segment-max to -inf
// =====================================================================
__global__ void init_layer_kernel(float* __restrict__ hn, float* __restrict__ denom,
                                  int* __restrict__ amax, float* __restrict__ few_sum, int H)
{
    const int i = blockIdx.x * blockDim.x + threadIdx.x;
    if (i < NN * HIDN) hn[i] = 0.0f;
    if (i < NN * H) { denom[i] = 0.0f; amax[i] = NEG_ORD; }
    if (i == 0) *few_sum = 0.0f;
}

// =====================================================================
// Edge weight MLP: few = ew * sigmoid(relu(ew*W1+b1)@W2 + b2); sum for mean
// =====================================================================
__global__ void edge_mlp_kernel(const float* __restrict__ ew,
                                const float* __restrict__ W1, const float* __restrict__ b1,
                                const float* __restrict__ W2, const float* __restrict__ b2s,
                                int layer, float* __restrict__ few, float* __restrict__ few_sum)
{
    const int e = blockIdx.x * blockDim.x + threadIdx.x;
    float f = 0.0f;
    if (e < EE) {
        const float w = ew[e];
        float s = b2s[layer];
#pragma unroll 8
        for (int j = 0; j < 32; ++j) {
            float hj = w * W1[layer * 32 + j] + b1[layer * 32 + j];
            hj = hj > 0.0f ? hj : 0.0f;
            s += hj * W2[layer * 32 + j];
        }
        const float sig = 1.0f / (1.0f + __expf(-s));
        f = w * sig;
        few[e] = f;
    }
    // wave32 reduction then one atomic per wave
    for (int off = 16; off > 0; off >>= 1) f += __shfl_xor(f, off, 32);
    if ((threadIdx.x & 31) == 0) atomicAdd(few_sum, f);
}

// a_edge = e_attr * dot(lin_edge_W[head], att_edge[head]) -- precompute per-head dot
__global__ void head_dot_kernel(const float* __restrict__ lin_edge_W,
                                const float* __restrict__ att_edge,
                                int layer, int H, float* __restrict__ le_dot)
{
    const int hh = threadIdx.x;
    if (hh < H) {
        const int C = HIDN / H;
        float s = 0.0f;
        for (int c = 0; c < C; ++c)
            s += lin_edge_W[layer * HIDN + hh * C + c] * att_edge[layer * HIDN + hh * C + c];
        le_dot[hh] = s;
    }
}

// =====================================================================
// Per-node attention coefficients a_src / a_dst
// =====================================================================
__global__ void node_att_kernel(const float* __restrict__ xl,
                                const float* __restrict__ att_src, const float* __restrict__ att_dst,
                                int layer, int H, float* __restrict__ a_src, float* __restrict__ a_dst)
{
    const int n = blockIdx.x * blockDim.x + threadIdx.x;
    if (n >= NN) return;
    const int C = HIDN / H;
    for (int hh = 0; hh < H; ++hh) {
        float s1 = 0.0f, s2 = 0.0f;
        for (int c = 0; c < C; ++c) {
            const float v = xl[(size_t)n * HIDN + hh * C + c];
            s1 += v * att_src[layer * HIDN + hh * C + c];
            s2 += v * att_dst[layer * HIDN + hh * C + c];
        }
        a_src[n * H + hh] = s1;
        a_dst[n * H + hh] = s2;
    }
}

// =====================================================================
// alpha = leaky_relu(a_src[src] + a_dst[dst] + e_attr*le_dot) ; segment max
// =====================================================================
__global__ void alpha_kernel(const int* __restrict__ ei, const float* __restrict__ few,
                             const float* __restrict__ few_sum, const float* __restrict__ le_dot,
                             const float* __restrict__ a_src, const float* __restrict__ a_dst,
                             int H, float* __restrict__ alpha, int* __restrict__ amax)
{
    const int t = blockIdx.x * blockDim.x + threadIdx.x;
    if (t >= (EE + NN) * H) return;
    const int j = t / H, hh = t - j * H;
    int s, d; float ea;
    if (j < EE) { s = ei[j]; d = ei[EE + j]; ea = few[j]; }
    else        { s = j - EE; d = s;         ea = *few_sum * (1.0f / (float)EE); }
    float al = a_src[s * H + hh] + a_dst[d * H + hh] + ea * le_dot[hh];
    al = al > 0.0f ? al : 0.2f * al;
    alpha[t] = al;
    atomicMax(&amax[d * H + hh], f2o(al));
}

// ex = exp(alpha - amax[dst]) ; segment sum into denom
__global__ void exp_kernel(const int* __restrict__ ei, int H, float* __restrict__ alpha,
                           const int* __restrict__ amax, float* __restrict__ denom)
{
    const int t = blockIdx.x * blockDim.x + threadIdx.x;
    if (t >= (EE + NN) * H) return;
    const int j = t / H, hh = t - j * H;
    const int d = (j < EE) ? ei[EE + j] : (j - EE);
    const float ex = __expf(alpha[t] - o2f(amax[d * H + hh]));
    alpha[t] = ex;
    atomicAdd(&denom[d * H + hh], ex);
}

// =====================================================================
// Aggregation: one wave per edge, 4 channels per lane, f32 atomics into hn
// =====================================================================
__global__ void aggr_kernel(const int* __restrict__ ei, const float* __restrict__ xl,
                            const float* __restrict__ ex, const float* __restrict__ denom,
                            int H, float* __restrict__ hn)
{
    const int gw   = (blockIdx.x * blockDim.x + threadIdx.x) >> 5;
    const int lane = threadIdx.x & 31;
    if (gw >= EE + NN) return;
    int s, d;
    if (gw < EE) { s = ei[gw]; d = ei[EE + gw]; }
    else         { s = gw - EE; d = s; }
    const int C = HIDN / H;
    __builtin_prefetch(&xl[(size_t)s * HIDN + lane], 0, 0);   // global_prefetch_b8
#pragma unroll
    for (int q = 0; q < 4; ++q) {
        const int ch = lane + q * 32;
        const int hh = ch / C;
        const float w = ex[gw * H + hh] / (denom[d * H + hh] + 1e-16f);
        atomicAdd(&hn[(size_t)d * HIDN + ch], w * xl[(size_t)s * HIDN + ch]);
    }
}

// h = elu(bn(hn + gat_bias))
__global__ void post_kernel(const float* __restrict__ hn, const float* __restrict__ gb,
                            const float* __restrict__ gg, const float* __restrict__ gbe,
                            const float* __restrict__ gm, const float* __restrict__ gv,
                            int layer, float* __restrict__ h)
{
    const int t = blockIdx.x * blockDim.x + threadIdx.x;
    if (t >= NN * HIDN) return;
    const int ch = t & (HIDN - 1);
    float v = hn[t] + gb[layer * HIDN + ch];
    v = (v - gm[layer * HIDN + ch]) * rsqrtf(gv[layer * HIDN + ch] + 1e-5f)
        * gg[layer * HIDN + ch] + gbe[layer * HIDN + ch];
    h[t] = v > 0.0f ? v : (__expf(v) - 1.0f);
}

__global__ void sym_kernel(float* __restrict__ h, const float* __restrict__ sym)
{
    const int t = blockIdx.x * blockDim.x + threadIdx.x;
    if (t < NN * HIDN) h[t] *= sym[t & (HIDN - 1)];
}

// c1 = bn(relu(c1))
__global__ void clf_bn_kernel(float* __restrict__ c1, const float* __restrict__ g,
                              const float* __restrict__ b, const float* __restrict__ m,
                              const float* __restrict__ v)
{
    const int t = blockIdx.x * blockDim.x + threadIdx.x;
    if (t >= NN * 64) return;
    const int ch = t & 63;
    float x = c1[t];
    x = x > 0.0f ? x : 0.0f;
    c1[t] = (x - m[ch]) * rsqrtf(v[ch] + 1e-5f) * g[ch] + b[ch];
}

__global__ void relu_kernel(float* __restrict__ p, int n)
{
    const int t = blockIdx.x * blockDim.x + threadIdx.x;
    if (t < n) { float x = p[t]; p[t] = x > 0.0f ? x : 0.0f; }
}

// logits = c2[Nx32] @ W3[32x2] + b3
__global__ void final_kernel(const float* __restrict__ c2, const float* __restrict__ W3,
                             const float* __restrict__ b3, float* __restrict__ out)
{
    const int n = blockIdx.x * blockDim.x + threadIdx.x;
    if (n >= NN) return;
    float s0 = b3[0], s1 = b3[1];
#pragma unroll 8
    for (int k = 0; k < 32; ++k) {
        const float v = c2[n * 32 + k];
        s0 += v * W3[k * 2 + 0];
        s1 += v * W3[k * 2 + 1];
    }
    out[n * 2 + 0] = s0;
    out[n * 2 + 1] = s1;
}

// =====================================================================
extern "C" void kernel_launch(void* const* d_in, const int* in_sizes, int n_in,
                              void* d_out, int out_size, void* d_ws, size_t ws_size,
                              hipStream_t stream)
{
    const float* x        = (const float*)d_in[0];
    const int*   ei       = (const int*)  d_in[1];
    const float* ew       = (const float*)d_in[2];
    const float* Wproj    = (const float*)d_in[3];
    const float* bproj    = (const float*)d_in[4];
    const float* gatW     = (const float*)d_in[5];
    const float* att_src  = (const float*)d_in[6];
    const float* att_dst  = (const float*)d_in[7];
    const float* att_edge = (const float*)d_in[8];
    const float* lin_edge = (const float*)d_in[9];
    const float* gat_bias = (const float*)d_in[10];
    const float* bn_g     = (const float*)d_in[11];
    const float* bn_b     = (const float*)d_in[12];
    const float* bn_m     = (const float*)d_in[13];
    const float* bn_v     = (const float*)d_in[14];
    const float* ew_W1    = (const float*)d_in[15];
    const float* ew_b1    = (const float*)d_in[16];
    const float* ew_W2    = (const float*)d_in[17];
    const float* ew_b2    = (const float*)d_in[18];
    const float* sym      = (const float*)d_in[19];
    const float* clf_W1   = (const float*)d_in[20];
    const float* clf_b1   = (const float*)d_in[21];
    const float* cbn_g    = (const float*)d_in[22];
    const float* cbn_b    = (const float*)d_in[23];
    const float* cbn_m    = (const float*)d_in[24];
    const float* cbn_v    = (const float*)d_in[25];
    const float* clf_W2   = (const float*)d_in[26];
    const float* clf_b2   = (const float*)d_in[27];
    const float* clf_W3   = (const float*)d_in[28];
    const float* clf_b3   = (const float*)d_in[29];
    float* out = (float*)d_out;

    // ---- workspace carve-out (floats) ----
    float* ws      = (float*)d_ws;
    float* h       = ws;                         // N*128
    float* xl      = h  + (size_t)NN * HIDN;     // N*128  (reused as c1 [N*64])
    float* hn      = xl + (size_t)NN * HIDN;     // N*128  (reused as c2 [N*32])
    float* alpha   = hn + (size_t)NN * HIDN;     // (E+N)*4 (alpha, then ex)
    float* a_src   = alpha + (size_t)(EE + NN) * 4;
    float* a_dst   = a_src + (size_t)NN * 4;
    float* denom   = a_dst + (size_t)NN * 4;
    int*   amax    = (int*)(denom + (size_t)NN * 4);
    float* few     = (float*)(amax + (size_t)NN * 4);
    float* few_sum = few + EE;
    float* le_dot  = few_sum + 1;

    const int HEADS[3] = {4, 4, 1};
    const int ROWB = (NN + 15) / 16;           // 6250 row-tiles (exact)

    // input projection: h = x @ Wproj + bproj
    gemm_wmma<64, 128><<<ROWB, 256, 0, stream>>>(x, Wproj, bproj, h, NN);

    for (int i = 0; i < 3; ++i) {
        const int H    = HEADS[i];
        const int totE = (EE + NN) * H;

        init_layer_kernel<<<(NN * HIDN + 255) / 256, 256, 0, stream>>>(hn, denom, amax, few_sum, H);
        edge_mlp_kernel<<<(EE + 255) / 256, 256, 0, stream>>>(ew, ew_W1, ew_b1, ew_W2, ew_b2, i, few, few_sum);
        head_dot_kernel<<<1, 32, 0, stream>>>(lin_edge, att_edge, i, H, le_dot);

        // xl = h @ gat_W[i]
        gemm_wmma<128, 128><<<ROWB, 256, 0, stream>>>(h, gatW + (size_t)i * HIDN * HIDN, nullptr, xl, NN);

        node_att_kernel<<<(NN + 255) / 256, 256, 0, stream>>>(xl, att_src, att_dst, i, H, a_src, a_dst);
        alpha_kernel<<<(totE + 255) / 256, 256, 0, stream>>>(ei, few, few_sum, le_dot, a_src, a_dst, H, alpha, amax);
        exp_kernel<<<(totE + 255) / 256, 256, 0, stream>>>(ei, H, alpha, amax, denom);
        aggr_kernel<<<((EE + NN) * 32 + 255) / 256, 256, 0, stream>>>(ei, xl, alpha, denom, H, hn);
        post_kernel<<<(NN * HIDN + 255) / 256, 256, 0, stream>>>(hn, gat_bias, bn_g, bn_b, bn_m, bn_v, i, h);
    }

    // classifier
    sym_kernel<<<(NN * HIDN + 255) / 256, 256, 0, stream>>>(h, sym);
    gemm_wmma<128, 64><<<ROWB, 128, 0, stream>>>(h, clf_W1, clf_b1, xl, NN);          // c1 in xl
    clf_bn_kernel<<<(NN * 64 + 255) / 256, 256, 0, stream>>>(xl, cbn_g, cbn_b, cbn_m, cbn_v);
    gemm_wmma<64, 32><<<ROWB, 64, 0, stream>>>(xl, clf_W2, clf_b2, hn, NN);           // c2 in hn
    relu_kernel<<<(NN * 32 + 255) / 256, 256, 0, stream>>>(hn, NN * 32);
    final_kernel<<<(NN + 255) / 256, 256, 0, stream>>>(hn, clf_W3, clf_b3, out);
}
